// BiMambaEncoder_21388937134866
// MI455X (gfx1250) — compile-verified, matrix operand order checked by reading, not run on hardware
//
#include <hip/hip_runtime.h>
#include <hip/hip_bf16.h>

// ---------------- problem constants ----------------
#define B_SZ    4
#define LSEQ    2048
#define DMODEL  256
#define DINNER  512
#define DSTATE  64
#define NHEADS  16
#define HEADDIM 32
#define CONVDIM 640           // D_INNER + 2*D_STATE
#define DINPROJ 1168          // 2*D_INNER + 2*D_STATE + NHEADS
#define DCONV   4
#define NLAYERS 8
#define CHUNKL  64
#define ROWS    (B_SZ * LSEQ) // 8192
#define EPSF    1e-5f

typedef __attribute__((ext_vector_type(16))) __bf16 v16bf;
typedef __attribute__((ext_vector_type(8)))  float  v8f;

static __device__ __forceinline__ float siluf(float x)     { return x / (1.f + __expf(-x)); }
static __device__ __forceinline__ float softplusf(float x) { return (x > 20.f) ? x : log1pf(__expf(x)); }

// ---------------------------------------------------------------------------
// Tensor Data Mover: 2D tile (fp32) from global -> LDS, one DMA per tile.
// Descriptor bitfields per CDNA5 ISA §8.3/8.4 (D# group0 128b, group1 256b).
// ---------------------------------------------------------------------------
#if defined(__has_builtin)
#if __has_builtin(__builtin_amdgcn_tensor_load_to_lds)
#define HAVE_TDM 1
#endif
#endif

#ifdef HAVE_TDM
typedef __attribute__((ext_vector_type(4))) unsigned int u32x4;
typedef __attribute__((ext_vector_type(4))) int          i32x4;
typedef __attribute__((ext_vector_type(8))) int          i32x8;

static __device__ __forceinline__ void tdm_load_2d_f32(
    unsigned lds_off, const void* gptr,
    unsigned tensor_d0, unsigned tensor_d1,
    unsigned tile_d0, unsigned tile_d1,
    unsigned long long stride0_elems) {
  const unsigned long long ga = (unsigned long long)gptr;
  u32x4 g0;
  g0[0] = 1u;                                               // count=1 (valid user D#)
  g0[1] = lds_off;                                          // lds_addr (bytes)
  g0[2] = (unsigned)(ga & 0xFFFFFFFFu);                     // global_addr[31:0]
  g0[3] = (unsigned)((ga >> 32) & 0x01FFFFFFu) | (2u << 30);// global_addr[56:32] | type=2
  i32x8 g1;
  g1[0] = (int)(2u << 16);                                  // data_size=2 (4B), mask=0
  g1[1] = (int)((tensor_d0 & 0xFFFFu) << 16);               // tensor_dim0[15:0]
  g1[2] = (int)((tensor_d0 >> 16) | ((tensor_d1 & 0xFFFFu) << 16));
  g1[3] = (int)(((tensor_d1 >> 16) & 0xFFFFu) | (tile_d0 << 16));
  g1[4] = (int)(tile_d1 & 0xFFFFu);                         // tile_dim1 | tile_dim2=0
  g1[5] = (int)(stride0_elems & 0xFFFFFFFFull);             // tensor_dim0_stride[31:0]
  g1[6] = (int)((stride0_elems >> 32) & 0xFFFFull);         // stride0[47:32] | stride1=0
  g1[7] = 0;
  const i32x4 z4 = {0, 0, 0, 0};
#if __clang_major__ >= 23
  const i32x8 z8 = {0, 0, 0, 0, 0, 0, 0, 0};
  __builtin_amdgcn_tensor_load_to_lds(g0, g1, z4, z4, z8, 0);
#else
  __builtin_amdgcn_tensor_load_to_lds(g0, g1, z4, z4, 0);
#endif
}
#endif // HAVE_TDM

// ---------------------------------------------------------------------------
// bf16 WMMA fragment loaders (fp32 source, convert per element).
// CDNA5 16-bit A 16x32 layout (ISA 7.12.2): lane group khalf=lane>>4 holds
// K = khalf*8 + [0..7] in VGPRs 0-3 and K = 16 + khalf*8 + [0..7] in VGPRs 4-7.
// B fragments (K x 16, weights stored (N,K) row-major) use the same pattern
// with row = output column. frag_ks takes an arbitrary K-stride so transposed
// LDS staging is never needed.
// ---------------------------------------------------------------------------
static __device__ __forceinline__ v16bf frag_ks(const float* rowPtr, int kstride,
                                                int kBase, int khalf) {
  v16bf f;
  const float* q = rowPtr + (size_t)(kBase + khalf * 8) * kstride;
#pragma unroll
  for (int e = 0; e < 8; ++e) {
    f[e]     = (__bf16)q[(size_t)e * kstride];
    f[e + 8] = (__bf16)q[(size_t)(16 + e) * kstride];
  }
  return f;
}

// ---------------------------------------------------------------------------
// GEMM: Out[m,n] = sum_k A[m,k] * W[n, wOff+k]  (+ Res[m,n])
// Each wave computes a 32x32 tile with v_wmma_f32_16x16x32_bf16; 8 waves/block
// cover a 256-row stripe. EXEC stays all-ones at every WMMA.
// ---------------------------------------------------------------------------
__global__ __launch_bounds__(256) void gemm_bf16_wmma(
    const float* __restrict__ A, int lda,
    const float* __restrict__ W, int ldw, int wOff,
    const float* __restrict__ Res,
    float* __restrict__ Out, int ldo,
    int M, int N, int K) {
  const int wave  = threadIdx.x >> 5;
  const int lane  = threadIdx.x & 31;
  const int mBase = blockIdx.y * 256 + wave * 32;
  const int nBase = blockIdx.x * 32;
  if (mBase >= M || nBase >= N) return;      // wave-uniform

  const int lm  = lane & 15;
  const int ld2 = lane >> 4;

  int ra0 = mBase + lm;       if (ra0 > M - 1) ra0 = M - 1;
  int ra1 = mBase + 16 + lm;  if (ra1 > M - 1) ra1 = M - 1;
  int rb0 = nBase + lm;       if (rb0 > N - 1) rb0 = N - 1;
  int rb1 = nBase + 16 + lm;  if (rb1 > N - 1) rb1 = N - 1;

  const float* pa0 = A + (size_t)ra0 * lda;
  const float* pa1 = A + (size_t)ra1 * lda;
  const float* pb0 = W + (size_t)rb0 * ldw + wOff;
  const float* pb1 = W + (size_t)rb1 * ldw + wOff;

  v8f c00 = {}, c01 = {}, c10 = {}, c11 = {};
  for (int k = 0; k < K; k += 32) {
    // speculative prefetch of the next K-slab (global_prefetch_b8; OOB is dropped)
    __builtin_prefetch(pa0 + k + 32, 0, 0);
    __builtin_prefetch(pb0 + k + 32, 0, 0);
    v16bf a0 = frag_ks(pa0, 1, k, ld2);
    v16bf a1 = frag_ks(pa1, 1, k, ld2);
    v16bf b0 = frag_ks(pb0, 1, k, ld2);
    v16bf b1 = frag_ks(pb1, 1, k, ld2);
    c00 = __builtin_amdgcn_wmma_f32_16x16x32_bf16(false, a0, false, b0, (short)0, c00, false, false);
    c01 = __builtin_amdgcn_wmma_f32_16x16x32_bf16(false, a0, false, b1, (short)0, c01, false, false);
    c10 = __builtin_amdgcn_wmma_f32_16x16x32_bf16(false, a1, false, b0, (short)0, c10, false, false);
    c11 = __builtin_amdgcn_wmma_f32_16x16x32_bf16(false, a1, false, b1, (short)0, c11, false, false);
  }

  // C/D layout: VGPR r -> M = (lane/16)*8 + r ; N = lane%16
#pragma unroll
  for (int mi = 0; mi < 2; ++mi) {
#pragma unroll
    for (int ni = 0; ni < 2; ++ni) {
      v8f c = (mi == 0) ? ((ni == 0) ? c00 : c01) : ((ni == 0) ? c10 : c11);
#pragma unroll
      for (int r = 0; r < 8; ++r) {
        int m = mBase + mi * 16 + ld2 * 8 + r;
        int n = nBase + ni * 16 + lm;
        if (m < M && n < N) {
          float v = c[r];
          if (Res) v += Res[(size_t)m * ldo + n];
          Out[(size_t)m * ldo + n] = v;
        }
      }
    }
  }
}

// ---------------------------------------------------------------------------
// LayerNorm over D_MODEL=256. One block per (b,l) row, one thread per element.
// ---------------------------------------------------------------------------
__global__ __launch_bounds__(256) void ln_kernel(const float* __restrict__ h,
                                                 const float* __restrict__ w,
                                                 const float* __restrict__ b,
                                                 float* __restrict__ out) {
  const int row = blockIdx.x, t = threadIdx.x;
  __shared__ float red[256];
  float x = h[(size_t)row * DMODEL + t];
  red[t] = x;
  __syncthreads();
  for (int s = 128; s > 0; s >>= 1) { if (t < s) red[t] += red[t + s]; __syncthreads(); }
  const float mu = red[0] * (1.f / DMODEL);
  __syncthreads();
  const float d = x - mu;
  red[t] = d * d;
  __syncthreads();
  for (int s = 128; s > 0; s >>= 1) { if (t < s) red[t] += red[t + s]; __syncthreads(); }
  const float var = red[0] * (1.f / DMODEL);
  out[(size_t)row * DMODEL + t] = d * rsqrtf(var + EPSF) * w[t] + b[t];
}

// ---------------------------------------------------------------------------
// Causal depthwise conv(4) + bias + SiLU over xBC channels, plus dt softplus.
// zx row layout: [0:512)=z  [512:1152)=xBC  [1152:1168)=dt_raw
// ---------------------------------------------------------------------------
__global__ __launch_bounds__(256) void conv_dt_kernel(
    const float* __restrict__ zx,
    const float* __restrict__ convw, const float* __restrict__ convb,
    const float* __restrict__ dtbias,
    float* __restrict__ xs, float* __restrict__ Bm, float* __restrict__ Cm,
    float* __restrict__ dt) {
  const size_t idx = (size_t)blockIdx.x * 256 + threadIdx.x;
  const size_t total = (size_t)ROWS * (CONVDIM + NHEADS);
  if (idx >= total) return;
  const int c = (int)(idx % (CONVDIM + NHEADS));
  const int l = (int)((idx / (CONVDIM + NHEADS)) % LSEQ);
  const int b = (int)(idx / ((size_t)(CONVDIM + NHEADS) * LSEQ));
  const size_t rowBase = (size_t)(b * LSEQ);
  if (c < CONVDIM) {
    float acc = convb[c];
#pragma unroll
    for (int k = 0; k < DCONV; ++k) {
      const int tt = l + k - (DCONV - 1);
      if (tt >= 0) acc += zx[(rowBase + tt) * DINPROJ + DINNER + c] * convw[c * DCONV + k];
    }
    const float v = siluf(acc);
    const size_t r = rowBase + l;
    if (c < DINNER)               xs[r * DINNER + c] = v;
    else if (c < DINNER + DSTATE) Bm[r * DSTATE + (c - DINNER)] = v;
    else                          Cm[r * DSTATE + (c - DINNER - DSTATE)] = v;
  } else {
    const int hh = c - CONVDIM;
    const float rr = zx[(rowBase + l) * DINPROJ + (DINNER + CONVDIM) + hh] + dtbias[hh];
    dt[(rowBase + l) * NHEADS + hh] = softplusf(rr);
  }
}

// ---------------------------------------------------------------------------
// SSD chunked scan, WMMA version. One workgroup (8 waves) per (batch, head);
// sequential over 32 chunks carrying state S[32][64] in LDS (~49 KB total).
// Chunk operand tiles (Bc 64x64, Xc 64x32) are staged by the Tensor Data
// Mover (one tensor_load_to_lds per tile from wave 0; TENSORcnt + barrier
// before use). All four chunk contractions run on v_wmma_f32_16x16x32_bf16:
//   CB   = Cc . Bc^T          (64x64, K=64)  -> att built in place (elementwise)
//   y    = att . Xc           (64x32, K=64)
//   y   += exp(cs_i) * Cc . S (64x32, K=64)  (S[p][n] is K-contiguous)
//   dS   = (w*X)^T . Bc       (32x64, K=64)  (A-fragment built on the fly)
// ---------------------------------------------------------------------------
__global__ __launch_bounds__(256) void ssd_kernel(
    const float* __restrict__ xs, const float* __restrict__ dt,
    const float* __restrict__ Bm, const float* __restrict__ Cm,
    const float* __restrict__ alog, const float* __restrict__ Dv,
    float* __restrict__ y) {
  const int b = blockIdx.x / NHEADS;
  const int h = blockIdx.x % NHEADS;
  const int t = threadIdx.x;
  const int wave = t >> 5, lane = t & 31;
  const int lm = lane & 15, ld2 = lane >> 4;

  __shared__ float S[HEADDIM][DSTATE];   // 8 KB  state [p][n]
  __shared__ float Bc[CHUNKL][DSTATE];   // 16 KB [j][n]
  __shared__ float Xc[CHUNKL][HEADDIM];  // 8 KB  [j][p]
  __shared__ float CB[CHUNKL][CHUNKL];   // 16 KB [i][j], reused as att in place
  __shared__ float csS[CHUNKL], dts[CHUNKL], wj[CHUNKL];

  const float A  = -__expf(alog[h]);
  const float Dh = Dv[h];
  const size_t rowBase = (size_t)(b * LSEQ);

#pragma unroll
  for (int r = 0; r < 8; ++r) { int i = t + r * 256; S[i >> 6][i & 63] = 0.f; }
  __syncthreads();

  for (int k = 0; k < LSEQ / CHUNKL; ++k) {
    const int l0 = k * CHUNKL;
    // ---- stage chunk operands in LDS ----
#ifdef HAVE_TDM
    if (t < 32) {
      // Bc tile: 64 rows x 64 elems, rows contiguous (stride 64)
      tdm_load_2d_f32((unsigned)(unsigned long long)(uintptr_t)&Bc[0][0],
                      Bm + (rowBase + l0) * DSTATE,
                      DSTATE, (unsigned)ROWS, DSTATE, CHUNKL, DSTATE);
      // Xc tile: 64 rows x 32 elems, row stride 512 (head slice of xs)
      tdm_load_2d_f32((unsigned)(unsigned long long)(uintptr_t)&Xc[0][0],
                      xs + (rowBase + l0) * DINNER + h * HEADDIM,
                      HEADDIM, (unsigned)ROWS, HEADDIM, CHUNKL, DINNER);
      __builtin_amdgcn_s_wait_tensorcnt(0);
    }
#else
#pragma unroll
    for (int r = 0; r < 16; ++r) {          // Bc: 4096 elems
      int i = (t + r * 256) >> 6, n = (t + r * 256) & 63;
      Bc[i][n] = Bm[(rowBase + l0 + i) * DSTATE + n];
    }
#pragma unroll
    for (int r = 0; r < 8; ++r) {           // Xc: 2048 elems
      int i = (t + r * 256) >> 5, p = (t + r * 256) & 31;
      Xc[i][p] = xs[(rowBase + l0 + i) * DINNER + h * HEADDIM + p];
    }
#endif
    if (t < CHUNKL) dts[t] = dt[(rowBase + l0 + t) * NHEADS + h];
    __syncthreads();
    if (t == 0) {                           // serial in-chunk scan (64 steps)
      float c = 0.f;
      for (int i = 0; i < CHUNKL; ++i) { c += dts[i] * A; csS[i] = c; }
      const float cl = csS[CHUNKL - 1];
      for (int j = 0; j < CHUNKL; ++j) wj[j] = __expf(cl - csS[j]) * dts[j];
    }
    __syncthreads();

    // ---- CB[i][j] = Cc[i,:] . Bc[j,:] : 16 tiles, 2 per wave ----
    for (int tt = wave; tt < 16; tt += 8) {
      const int i0 = (tt >> 2) * 16, j0 = (tt & 3) * 16;
      v8f acc = {};
#pragma unroll
      for (int ks = 0; ks < DSTATE; ks += 32) {
        v16bf a = frag_ks(Cm + (rowBase + l0 + i0 + lm) * DSTATE, 1, ks, ld2);
        v16bf w = frag_ks(&Bc[j0 + lm][0], 1, ks, ld2);
        acc = __builtin_amdgcn_wmma_f32_16x16x32_bf16(false, a, false, w, (short)0, acc, false, false);
      }
#pragma unroll
      for (int r = 0; r < 8; ++r) CB[i0 + ld2 * 8 + r][j0 + lm] = acc[r];
    }
    __syncthreads();

    // ---- att in place: att[i][j] = (j<=i) ? CB*exp(cs_i-cs_j)*dt_j : 0 ----
#pragma unroll
    for (int r = 0; r < 16; ++r) {
      int idx = t + r * 256;
      int i = idx >> 6, j = idx & 63;
      float v = (j <= i) ? CB[i][j] * __expf(csS[i] - csS[j]) * dts[j] : 0.f;
      CB[i][j] = v;
    }
    __syncthreads();

    // ---- y tile per wave: out 64(i) x 32(p) = 4x2 tiles ----
    {
      const int i0 = (wave >> 1) * 16, p0 = (wave & 1) * 16;
      v8f accA = {}, accI = {};
#pragma unroll
      for (int ks = 0; ks < CHUNKL; ks += 32) {
        v16bf a = frag_ks(&CB[i0 + lm][0], 1, ks, ld2);          // att[i][j]
        v16bf w = frag_ks(&Xc[0][p0 + lm], HEADDIM, ks, ld2);    // X[j][p] as W[p][j]
        accA = __builtin_amdgcn_wmma_f32_16x16x32_bf16(false, a, false, w, (short)0, accA, false, false);
      }
#pragma unroll
      for (int ks = 0; ks < DSTATE; ks += 32) {
        v16bf a = frag_ks(Cm + (rowBase + l0 + i0 + lm) * DSTATE, 1, ks, ld2);
        v16bf w = frag_ks(&S[p0 + lm][0], 1, ks, ld2);           // S[p][n], K=n
        accI = __builtin_amdgcn_wmma_f32_16x16x32_bf16(false, a, false, w, (short)0, accI, false, false);
      }
#pragma unroll
      for (int r = 0; r < 8; ++r) {
        const int i = i0 + ld2 * 8 + r;
        const int p = p0 + lm;
        const float v = accA[r] + accI[r] * __expf(csS[i]) + Dh * Xc[i][p];
        y[(rowBase + l0 + i) * DINNER + h * HEADDIM + p] = v;
      }
    }
    __syncthreads();   // all S reads done before S update

    // ---- state update: S[p][n] = exp(cs_last)*S + (w*X)^T . Bc ----
    {
      const int p0 = (wave >> 2) * 16, n0 = (wave & 3) * 16;
      const float gl = __expf(csS[CHUNKL - 1]);
      v8f acc = {};
#pragma unroll
      for (int ks = 0; ks < CHUNKL; ks += 32) {
        // A-fragment built on the fly: A[p][j] = wj[j] * Xc[j][p]
        v16bf a;
        const int p = p0 + lm;
#pragma unroll
        for (int e = 0; e < 8; ++e) {
          const int ja = ks + ld2 * 8 + e;
          const int jb = ks + 16 + ld2 * 8 + e;
          a[e]     = (__bf16)(wj[ja] * Xc[ja][p]);
          a[e + 8] = (__bf16)(wj[jb] * Xc[jb][p]);
        }
        v16bf w = frag_ks(&Bc[0][n0 + lm], DSTATE, ks, ld2);     // Bc[j][n] as W[n][j]
        acc = __builtin_amdgcn_wmma_f32_16x16x32_bf16(false, a, false, w, (short)0, acc, false, false);
      }
#pragma unroll
      for (int r = 0; r < 8; ++r) {
        const int p = p0 + ld2 * 8 + r;
        const int n = n0 + lm;
        S[p][n] = gl * S[p][n] + acc[r];
      }
    }
    __syncthreads();
  }
}

// ---------------------------------------------------------------------------
// y = y * silu(z); RMS-norm over 512; * norm_w. In place on y.
// ---------------------------------------------------------------------------
__global__ __launch_bounds__(256) void gate_rms_kernel(float* __restrict__ y,
                                                       const float* __restrict__ zx,
                                                       const float* __restrict__ nw) {
  const int row = blockIdx.x, t = threadIdx.x;
  __shared__ float red[256];
  const size_t yb = (size_t)row * DINNER;
  const size_t zb = (size_t)row * DINPROJ;
  float v0 = y[yb + t]       * siluf(zx[zb + t]);
  float v1 = y[yb + 256 + t] * siluf(zx[zb + 256 + t]);
  red[t] = v0 * v0 + v1 * v1;
  __syncthreads();
  for (int s = 128; s > 0; s >>= 1) { if (t < s) red[t] += red[t + s]; __syncthreads(); }
  const float rms = rsqrtf(red[0] * (1.f / DINNER) + EPSF);
  y[yb + t]       = v0 * rms * nw[t];
  y[yb + 256 + t] = v1 * rms * nw[256 + t];
}

// ---------------------------------------------------------------------------
// dst[b,l,:] = src[b, flip ? L-1-l : l, :]   (D_MODEL channels)
// ---------------------------------------------------------------------------
__global__ __launch_bounds__(256) void copy_rows_kernel(float* __restrict__ dst,
                                                        const float* __restrict__ src,
                                                        int flip) {
  const size_t idx = (size_t)blockIdx.x * 256 + threadIdx.x;
  if (idx >= (size_t)ROWS * DMODEL) return;
  const int d = (int)(idx % DMODEL);
  const int l = (int)((idx / DMODEL) % LSEQ);
  const int b = (int)(idx / ((size_t)DMODEL * LSEQ));
  const int ls = flip ? (LSEQ - 1 - l) : l;
  dst[idx] = src[((size_t)(b * LSEQ) + ls) * DMODEL + d];
}

// ---------------------------------------------------------------------------
extern "C" void kernel_launch(void* const* d_in, const int* in_sizes, int n_in,
                              void* d_out, int out_size, void* d_ws, size_t ws_size,
                              hipStream_t stream) {
  (void)in_sizes; (void)n_in; (void)out_size; (void)ws_size;
  const float* x = (const float*)d_in[0];
  float* out = (float*)d_out;

  // workspace carve-out (~110 MB fp32)
  float* ws  = (float*)d_ws;
  float* hF  = ws;
  float* hB  = hF  + (size_t)ROWS * DMODEL;
  float* hbf = hB  + (size_t)ROWS * DMODEL;
  float* lnb = hbf + (size_t)ROWS * DMODEL;
  float* zx  = lnb + (size_t)ROWS * DMODEL;
  float* xsb = zx  + (size_t)ROWS * DINPROJ;
  float* Bmb = xsb + (size_t)ROWS * DINNER;
  float* Cmb = Bmb + (size_t)ROWS * DSTATE;
  float* dtb = Cmb + (size_t)ROWS * DSTATE;
  float* yb  = dtb + (size_t)ROWS * NHEADS;

  const int copyBlocks = (int)(((size_t)ROWS * DMODEL + 255) / 256);
  copy_rows_kernel<<<copyBlocks, 256, 0, stream>>>(hF, x, 0);
  copy_rows_kernel<<<copyBlocks, 256, 0, stream>>>(hB, x, 1);

  for (int dir = 0; dir < 2; ++dir) {
    const int base = 1 + dir * 10;
    const float* lnw    = (const float*)d_in[base + 0];
    const float* lnbias = (const float*)d_in[base + 1];
    const float* win    = (const float*)d_in[base + 2];
    const float* convw  = (const float*)d_in[base + 3];
    const float* convb  = (const float*)d_in[base + 4];
    const float* alog   = (const float*)d_in[base + 5];
    const float* dtbias = (const float*)d_in[base + 6];
    const float* Dp     = (const float*)d_in[base + 7];
    const float* nw     = (const float*)d_in[base + 8];
    const float* wout   = (const float*)d_in[base + 9];
    float* h = dir ? hB : hF;

    for (int i = 0; i < NLAYERS; ++i) {
      ln_kernel<<<ROWS, 256, 0, stream>>>(h, lnw + i * DMODEL, lnbias + i * DMODEL, lnb);

      gemm_bf16_wmma<<<dim3((DINPROJ + 31) / 32, ROWS / 256), 256, 0, stream>>>(
          lnb, DMODEL, win + (size_t)i * DINPROJ * DMODEL, DMODEL, 0,
          nullptr, zx, DINPROJ, ROWS, DINPROJ, DMODEL);

      const int convThreads = ROWS * (CONVDIM + NHEADS);
      conv_dt_kernel<<<(convThreads + 255) / 256, 256, 0, stream>>>(
          zx, convw + (size_t)i * CONVDIM * DCONV, convb + (size_t)i * CONVDIM,
          dtbias + (size_t)i * NHEADS, xsb, Bmb, Cmb, dtb);

      ssd_kernel<<<B_SZ * NHEADS, 256, 0, stream>>>(
          xsb, dtb, Bmb, Cmb, alog + (size_t)i * NHEADS, Dp + (size_t)i * NHEADS, yb);

      gate_rms_kernel<<<ROWS, 256, 0, stream>>>(yb, zx, nw + (size_t)i * DINNER);

      // h += y @ Wout^T   (residual accumulated in GEMM epilogue)
      gemm_bf16_wmma<<<dim3(DMODEL / 32, ROWS / 256), 256, 0, stream>>>(
          yb, DINNER, wout + (size_t)i * DMODEL * DINNER, DINNER, 0,
          h, h, DMODEL, ROWS, DMODEL, DINNER);
    }
  }

  // hbf = time-reversed backward-stack output
  copy_rows_kernel<<<copyBlocks, 256, 0, stream>>>(hbf, hB, 1);

  // fusion: out = hF @ Wf[:, :256]^T + hbf @ Wf[:, 256:]^T
  const float* fw = (const float*)d_in[21];
  gemm_bf16_wmma<<<dim3(DMODEL / 32, ROWS / 256), 256, 0, stream>>>(
      hF, DMODEL, fw, 2 * DMODEL, 0, nullptr, out, DMODEL, ROWS, DMODEL, DMODEL);
  gemm_bf16_wmma<<<dim3(DMODEL / 32, ROWS / 256), 256, 0, stream>>>(
      hbf, DMODEL, fw, 2 * DMODEL, DMODEL, out, out, DMODEL, ROWS, DMODEL, DMODEL);
}
